// AnomalyTransformer_freq_29532195127322
// MI455X (gfx1250) — compile-verified, matrix operand order checked by reading, not run on hardware
//
#include <hip/hip_runtime.h>
#include <hip/hip_bf16.h>

// ---------------------------------------------------------------------------
// Model constants (match reference)
// ---------------------------------------------------------------------------
constexpr int Bb = 128, Ls = 100, CIN = 55, COUT = 55;
constexpr int Dd = 512, Hh = 8, DFF = 512, MODES = 32, Ee = 64;
constexpr long BLD    = (long)Bb * Ls * Dd;   // 6,553,600
constexpr long M_ROWS = (long)Bb * Ls;        // 12,800

typedef __attribute__((ext_vector_type(16))) __bf16 v16bf;
typedef __attribute__((ext_vector_type(8)))  float  v8f;

__device__ __forceinline__ int clampi(int v, int lo, int hi) {
    return v < lo ? lo : (v > hi ? hi : v);
}
__device__ __forceinline__ float gelu_exact(float x) {
    return 0.5f * x * (1.0f + erff(x * 0.7071067811865476f));
}
__device__ __forceinline__ unsigned short f2bf(float f) {
    unsigned int u = __float_as_uint(f);
    u += 0x7FFFu + ((u >> 16) & 1u);   // round-to-nearest-even
    return (unsigned short)(u >> 16);
}

// ---------------------------------------------------------------------------
// fp32 -> bf16 conversion (activations: exact size)
// ---------------------------------------------------------------------------
__global__ void cvt_bf16_kernel(const float* __restrict__ in,
                                unsigned short* __restrict__ out, long n) {
    long i = (long)blockIdx.x * 256 + threadIdx.x;
    if (i < n) out[i] = f2bf(in[i]);
}

// fp32 -> bf16 conversion for weights, zero-padded to Npad rows so the GEMM
// B-fragment loads are unconditional (no exec-mask branching in the k-loop).
__global__ void cvt_bf16_pad_kernel(const float* __restrict__ in,
                                    unsigned short* __restrict__ out,
                                    int N, int K, int Npad) {
    long i = (long)blockIdx.x * 256 + threadIdx.x;
    if (i >= (long)Npad * K) return;
    int row = (int)(i / K);
    out[i] = (row < N) ? f2bf(in[i]) : (unsigned short)0;
}

__global__ void zero_kernel(float* __restrict__ p, long n) {
    long i = (long)blockIdx.x * 256 + threadIdx.x;
    if (i < n) p[i] = 0.0f;
}

// ---------------------------------------------------------------------------
// WMMA GEMM: Y[M,N] = act( X[M,K] * W[N,K]^T + bias[N] + resid[M,N] )
// bf16 inputs, f32 accumulate via v_wmma_f32_16x16x32_bf16.
// Each wave32 computes a 16x64 output strip (4 independent accumulators):
// one A fragment feeds 4 WMMAs per 32-wide k-step. All fragment loads are
// contiguous 16-byte (b128) loads; W is pre-padded to a multiple of 64 rows
// so no bounds checks exist in the inner loop.
// M must be a multiple of 16 (M=12800 here); K a multiple of 32 (512 here).
// ---------------------------------------------------------------------------
__global__ __launch_bounds__(256) void gemm_wmma_kernel(
    const unsigned short* __restrict__ A,   // M x K bf16
    const unsigned short* __restrict__ W,   // Npad x K bf16 (zero-padded)
    const float* __restrict__ bias,         // N or null
    const float* __restrict__ resid,        // M x N or null
    float* __restrict__ Y,                  // M x N
    int M, int N, int K, int act)
{
    const int wave = threadIdx.x >> 5;
    const int lane = threadIdx.x & 31;
    const int nsup = (N + 63) >> 6;          // 64-col supertiles
    const long tile = (long)blockIdx.x * 8 + wave;
    const int tm = (int)(tile / nsup) * 16;
    const int tn = (int)(tile % nsup) * 64;
    if (tm >= M) return;                     // wave-uniform exit

    const int half = lane >> 4;              // K-half selector (ISA fragment layout)
    const int idx  = lane & 15;              // A row / B col within 16x16 tile
    // Per-lane base pointers; fragment = two contiguous 8-halfword runs:
    //   [k0 + half*8, +8) and [k0 + 16 + half*8, +8)
    const unsigned short* ar = A + (long)(tm + idx) * K + half * 8;
    const unsigned short* w0 = W + (long)(tn + idx) * K + half * 8;
    const unsigned short* w1 = w0 + (long)16 * K;
    const unsigned short* w2 = w0 + (long)32 * K;
    const unsigned short* w3 = w0 + (long)48 * K;

    v8f acc0 = {}, acc1 = {}, acc2 = {}, acc3 = {};
    union Frag { uint4 q[2]; v16bf v; };

#pragma unroll 4
    for (int k0 = 0; k0 < K; k0 += 32) {
        Frag fa, f0, f1, f2, f3;
        fa.q[0] = *(const uint4*)(ar + k0);
        fa.q[1] = *(const uint4*)(ar + k0 + 16);
        f0.q[0] = *(const uint4*)(w0 + k0);
        f0.q[1] = *(const uint4*)(w0 + k0 + 16);
        f1.q[0] = *(const uint4*)(w1 + k0);
        f1.q[1] = *(const uint4*)(w1 + k0 + 16);
        f2.q[0] = *(const uint4*)(w2 + k0);
        f2.q[1] = *(const uint4*)(w2 + k0 + 16);
        f3.q[0] = *(const uint4*)(w3 + k0);
        f3.q[1] = *(const uint4*)(w3 + k0 + 16);
        acc0 = __builtin_amdgcn_wmma_f32_16x16x32_bf16(false, fa.v, false, f0.v,
                                                       (short)0, acc0, false, false);
        acc1 = __builtin_amdgcn_wmma_f32_16x16x32_bf16(false, fa.v, false, f1.v,
                                                       (short)0, acc1, false, false);
        acc2 = __builtin_amdgcn_wmma_f32_16x16x32_bf16(false, fa.v, false, f2.v,
                                                       (short)0, acc2, false, false);
        acc3 = __builtin_amdgcn_wmma_f32_16x16x32_bf16(false, fa.v, false, f3.v,
                                                       (short)0, acc3, false, false);
    }

    // Epilogue: C/D layout -> lane holds col n, rows m = half*8 + v
    v8f accs[4] = {acc0, acc1, acc2, acc3};
#pragma unroll
    for (int j = 0; j < 4; ++j) {
        const int n = tn + j * 16 + idx;
        const float bv = (bias && n < N) ? bias[n] : 0.0f;
#pragma unroll
        for (int v = 0; v < 8; ++v) {
            const int m = tm + half * 8 + v;
            if (n < N) {
                float val = accs[j][v] + bv;
                if (resid) val += resid[(long)m * N + n];
                if (act) val = gelu_exact(val);
                Y[(long)m * N + n] = val;
            }
        }
    }
}

// ---------------------------------------------------------------------------
// Batched WMMA attention scores: S[b,h,l,s] = (1/8) * Q[b,l,h,:] . K[b,s,h,:]
// Per (b,h): a 100x100x64 GEMM with row stride D. Rows are clamped (not
// branched) for the 100->112 padding; stores are guarded. K=64 -> the k-loop
// fully unrolls into 8 WMMAs per wave.
// ---------------------------------------------------------------------------
__global__ __launch_bounds__(256) void scores_wmma_kernel(
    const unsigned short* __restrict__ Qbf,   // (B*L, D) bf16
    const unsigned short* __restrict__ Kbf,   // (B*L, D) bf16
    float* __restrict__ S)                    // (B,H,L,L)
{
    constexpr int TM = 7;   // ceil(100/16) row tiles
    constexpr int TN = 2;   // ceil(100/64) 64-col supertiles
    const int wave = threadIdx.x >> 5;
    const int lane = threadIdx.x & 31;
    const long tile = (long)blockIdx.x * 8 + wave;
    if (tile >= (long)Bb * Hh * TM * TN) return;   // wave-uniform exit
    const int tni = (int)(tile % TN);
    const int tmi = (int)((tile / TN) % TM);
    const int bh  = (int)(tile / (TN * TM));
    const int h = bh % Hh, b = bh / Hh;
    const int tm = tmi * 16, tn = tni * 64;

    const int half = lane >> 4;
    const int idx  = lane & 15;
    const long rowbase = (long)b * Ls;
    const int hoff = h * Ee + half * 8;
    const unsigned short* ar =
        Qbf + (rowbase + clampi(tm + idx, 0, Ls - 1)) * Dd + hoff;
    const unsigned short* w0 =
        Kbf + (rowbase + clampi(tn + idx, 0, Ls - 1)) * Dd + hoff;
    const unsigned short* w1 =
        Kbf + (rowbase + clampi(tn + 16 + idx, 0, Ls - 1)) * Dd + hoff;
    const unsigned short* w2 =
        Kbf + (rowbase + clampi(tn + 32 + idx, 0, Ls - 1)) * Dd + hoff;
    const unsigned short* w3 =
        Kbf + (rowbase + clampi(tn + 48 + idx, 0, Ls - 1)) * Dd + hoff;

    v8f acc0 = {}, acc1 = {}, acc2 = {}, acc3 = {};
    union Frag { uint4 q[2]; v16bf v; };
#pragma unroll
    for (int k0 = 0; k0 < Ee; k0 += 32) {
        Frag fa, f0, f1, f2, f3;
        fa.q[0] = *(const uint4*)(ar + k0);
        fa.q[1] = *(const uint4*)(ar + k0 + 16);
        f0.q[0] = *(const uint4*)(w0 + k0);
        f0.q[1] = *(const uint4*)(w0 + k0 + 16);
        f1.q[0] = *(const uint4*)(w1 + k0);
        f1.q[1] = *(const uint4*)(w1 + k0 + 16);
        f2.q[0] = *(const uint4*)(w2 + k0);
        f2.q[1] = *(const uint4*)(w2 + k0 + 16);
        f3.q[0] = *(const uint4*)(w3 + k0);
        f3.q[1] = *(const uint4*)(w3 + k0 + 16);
        acc0 = __builtin_amdgcn_wmma_f32_16x16x32_bf16(false, fa.v, false, f0.v,
                                                       (short)0, acc0, false, false);
        acc1 = __builtin_amdgcn_wmma_f32_16x16x32_bf16(false, fa.v, false, f1.v,
                                                       (short)0, acc1, false, false);
        acc2 = __builtin_amdgcn_wmma_f32_16x16x32_bf16(false, fa.v, false, f2.v,
                                                       (short)0, acc2, false, false);
        acc3 = __builtin_amdgcn_wmma_f32_16x16x32_bf16(false, fa.v, false, f3.v,
                                                       (short)0, acc3, false, false);
    }

    v8f accs[4] = {acc0, acc1, acc2, acc3};
#pragma unroll
    for (int j = 0; j < 4; ++j) {
        const int n = tn + j * 16 + idx;
#pragma unroll
        for (int v = 0; v < 8; ++v) {
            const int m = tm + half * 8 + v;
            if (m < Ls && n < Ls)
                S[((long)bh * Ls + m) * Ls + n] = accs[j][v] * 0.125f;
        }
    }
}

// In-place row softmax over the last axis of S (B,H,L,L)
__global__ __launch_bounds__(128) void softmax_inplace_kernel(float* __restrict__ S) {
    int bhl = blockIdx.x;
    int t = threadIdx.x;
    __shared__ float sm[128];
    float* row = S + (long)bhl * Ls;
    float v = (t < Ls) ? row[t] : -3.0e38f;
    sm[t] = v; __syncthreads();
    for (int s = 64; s > 0; s >>= 1) { if (t < s) sm[t] = fmaxf(sm[t], sm[t + s]); __syncthreads(); }
    float mx = sm[0]; __syncthreads();
    float ex = (t < Ls) ? expf(v - mx) : 0.0f;
    sm[t] = ex; __syncthreads();
    for (int s = 64; s > 0; s >>= 1) { if (t < s) sm[t] += sm[t + s]; __syncthreads(); }
    float inv = 1.0f / sm[0];
    if (t < Ls) row[t] = ex * inv;
}

// ---------------------------------------------------------------------------
// DFT trig table: trig[m*L + l] = (cos(2*pi*m*l/L), sin(2*pi*m*l/L))
// ---------------------------------------------------------------------------
__global__ void trig_kernel(float2* __restrict__ trig) {
    int i = blockIdx.x * 256 + threadIdx.x;
    if (i >= MODES * Ls) return;
    int l = i % Ls, m = i / Ls;
    float th = 6.283185307179586f * (float)(m * l) / (float)Ls;
    trig[i] = make_float2(cosf(th), sinf(th));
}

// xmean[b,c] = mean over l of x[b,l,c]
__global__ void colmean_kernel(const float* __restrict__ x, float* __restrict__ xm) {
    int i = blockIdx.x * 256 + threadIdx.x;
    if (i >= Bb * CIN) return;
    int c = i % CIN, b = i / CIN;
    float s = 0.0f;
    for (int l = 0; l < Ls; ++l) s += x[((long)b * Ls + l) * CIN + c];
    xm[i] = s * (1.0f / Ls);
}

// Circular conv3 embedding (optionally subtracting per-(b,c) mean) + positional emb
__global__ void embed_kernel(const float* __restrict__ x, const float* __restrict__ W,
                             const float* __restrict__ xmean, float* __restrict__ out) {
    long i = (long)blockIdx.x * 256 + threadIdx.x;
    if (i >= BLD) return;
    int d = (int)(i % Dd);
    int l = (int)((i / Dd) % Ls);
    int b = (int)(i / ((long)Dd * Ls));
    const float* Wd = W + (long)d * CIN * 3;
    float acc = 0.0f;
    for (int j = 0; j < 3; ++j) {
        int lm = l + j - 1;
        if (lm < 0) lm += Ls; else if (lm >= Ls) lm -= Ls;
        const float* xr = x + ((long)b * Ls + lm) * CIN;
        for (int c = 0; c < CIN; ++c) {
            float xv = xr[c];
            if (xmean) xv -= xmean[b * CIN + c];
            acc += xv * Wd[c * 3 + j];
        }
    }
    int hi = d >> 1;
    float div = expf(-(float)(2 * hi) * (9.210340371976184f / (float)Dd));
    float ang = (float)l * div;
    acc += (d & 1) ? cosf(ang) : sinf(ang);
    out[i] = acc;
}

// sigma[b,h,l] = 3^(sigmoid(5*(enc . sw[h] + sb[h])) + 1e-5) - 1
__global__ void sigma_kernel(const float* __restrict__ enc, const float* __restrict__ sw,
                             const float* __restrict__ sb, float* __restrict__ out) {
    int i = blockIdx.x * 256 + threadIdx.x;
    if (i >= Bb * Hh * Ls) return;
    int l = i % Ls, h = (i / Ls) % Hh, b = i / (Ls * Hh);
    const float* er = enc + ((long)b * Ls + l) * Dd;
    const float* wr = sw + (long)h * Dd;
    float acc = sb[h];
    for (int d = 0; d < Dd; ++d) acc += er[d] * wr[d];
    float sg = 1.0f / (1.0f + expf(-5.0f * acc));
    out[i] = powf(3.0f, sg + 1e-5f) - 1.0f;
}

// prior[b,h,l,s] = N(|l-s|; sigma[b,h,l])
__global__ void prior_kernel(const float* __restrict__ sigma, float* __restrict__ out) {
    long i = (long)blockIdx.x * 256 + threadIdx.x;
    if (i >= (long)Bb * Hh * Ls * Ls) return;
    int s = (int)(i % Ls);
    int l = (int)((i / Ls) % Ls);
    long bh = i / ((long)Ls * Ls);
    float sg = sigma[bh * Ls + l];
    float d = fabsf((float)(l - s));
    out[i] = expf(-d * d / (2.0f * sg * sg)) * 0.3989422804014327f / sg;
}

// Forward DFT over L (first MODES complex frequencies): xf[b,h,e,m]
__global__ void dft_fwd_kernel(const float* __restrict__ q, const float2* __restrict__ trig,
                               float* __restrict__ xf) {
    long i = (long)blockIdx.x * 256 + threadIdx.x;
    if (i >= (long)Bb * Hh * Ee * MODES) return;
    int m = (int)(i % MODES);
    int e = (int)((i / MODES) % Ee);
    int h = (int)((i / ((long)MODES * Ee)) % Hh);
    int b = (int)(i / ((long)MODES * Ee * Hh));
    const float* src = q + ((long)b * Ls) * Dd + h * Ee + e;
    const float2* tb = trig + m * Ls;
    float re = 0.0f, im = 0.0f;
    for (int l = 0; l < Ls; ++l) {
        float v = src[(long)l * Dd];
        re += v * tb[l].x;
        im -= v * tb[l].y;
    }
    xf[2 * i] = re; xf[2 * i + 1] = im;
}

// sel[b,h,o,m] = sum_e xf[b,h,e,m] * (wr + i wi)[h,e,o,m]   (complex)
__global__ void mode_gemm_kernel(const float* __restrict__ xf, const float* __restrict__ wr,
                                 const float* __restrict__ wi, float* __restrict__ sel) {
    long i = (long)blockIdx.x * 256 + threadIdx.x;
    if (i >= (long)Bb * Hh * Ee * MODES) return;
    int m = (int)(i % MODES);
    int o = (int)((i / MODES) % Ee);
    int h = (int)((i / ((long)MODES * Ee)) % Hh);
    int b = (int)(i / ((long)MODES * Ee * Hh));
    const float* xb  = xf + (((long)(b * Hh + h) * Ee) * MODES + m) * 2;
    const float* wrb = wr + ((long)h * Ee * Ee + o) * MODES + m;
    const float* wib = wi + ((long)h * Ee * Ee + o) * MODES + m;
    float re = 0.0f, im = 0.0f;
    for (int e = 0; e < Ee; ++e) {
        float ar = xb[(long)e * MODES * 2], ai = xb[(long)e * MODES * 2 + 1];
        float br = wrb[(long)e * Ee * MODES], bi = wib[(long)e * Ee * MODES];
        re += ar * br - ai * bi;
        im += ar * bi + ai * br;
    }
    sel[2 * i] = re; sel[2 * i + 1] = im;
}

// irfft with only first MODES modes set: out[b,l,h,o]
__global__ void idft_kernel(const float* __restrict__ sel, const float2* __restrict__ trig,
                            float* __restrict__ out, float scale) {
    long i = (long)blockIdx.x * 256 + threadIdx.x;
    if (i >= BLD) return;
    int o = (int)(i % Ee);
    int h = (int)((i / Ee) % Hh);
    int l = (int)((i / Dd) % Ls);
    int b = (int)(i / ((long)Dd * Ls));
    const float* sp = sel + (((long)(b * Hh + h) * Ee + o) * MODES) * 2;
    float acc = sp[0];                       // m = 0, coeff 1
    for (int m = 1; m < MODES; ++m) {        // hermitian doubling for 1 <= m < n/2
        float2 cs = trig[m * Ls + l];
        acc += 2.0f * (sp[2 * m] * cs.x - sp[2 * m + 1] * cs.y);
    }
    out[i] = acc * scale;
}

// qk[b,h,x,y] = ctanh( sum_e qf[b,h,e,x] * kf[b,h,e,y] )
__global__ void cross_qk_kernel(const float* __restrict__ qf, const float* __restrict__ kf,
                                float* __restrict__ qk) {
    long i = (long)blockIdx.x * 256 + threadIdx.x;
    if (i >= (long)Bb * Hh * MODES * MODES) return;
    int y  = (int)(i % MODES);
    int xm = (int)((i / MODES) % MODES);
    long bh = i / ((long)MODES * MODES);
    const float* qp = qf + ((bh * Ee) * MODES + xm) * 2;
    const float* kp = kf + ((bh * Ee) * MODES + y) * 2;
    float re = 0.0f, im = 0.0f;
    for (int e = 0; e < Ee; ++e) {
        float ar = qp[(long)e * MODES * 2], ai = qp[(long)e * MODES * 2 + 1];
        float br = kp[(long)e * MODES * 2], bi = kp[(long)e * MODES * 2 + 1];
        re += ar * br - ai * bi;
        im += ar * bi + ai * br;
    }
    // complex tanh: (sinh 2a + i sin 2b) / (cosh 2a + cos 2b), stabilized
    float A = 2.0f * re, Bv = 2.0f * im, reo, imo;
    if (fabsf(A) > 30.0f) { reo = (A > 0.0f) ? 1.0f : -1.0f; imo = 0.0f; }
    else {
        float den = coshf(A) + cosf(Bv);
        reo = sinhf(A) / den;
        imo = sinf(Bv) / den;
    }
    qk[2 * i] = reo; qk[2 * i + 1] = imo;
}

// qkv[b,h,e,x] = sum_y qk[b,h,x,y] * kf[b,h,e,y]
__global__ void cross_qkv_kernel(const float* __restrict__ qk, const float* __restrict__ kf,
                                 float* __restrict__ qkv) {
    long i = (long)blockIdx.x * 256 + threadIdx.x;
    if (i >= (long)Bb * Hh * Ee * MODES) return;
    int xm = (int)(i % MODES);
    int e  = (int)((i / MODES) % Ee);
    long bh = i / ((long)MODES * Ee);
    const float* qkb = qk + ((bh * MODES + xm) * MODES) * 2;
    const float* kp  = kf + ((bh * Ee + e) * MODES) * 2;
    float re = 0.0f, im = 0.0f;
    for (int y = 0; y < MODES; ++y) {
        float ar = qkb[2 * y], ai = qkb[2 * y + 1];
        float br = kp[2 * y],  bi = kp[2 * y + 1];
        re += ar * br - ai * bi;
        im += ar * bi + ai * br;
    }
    qkv[2 * i] = re; qkv[2 * i + 1] = im;
}

// series_decomp_multi: season = x - mixed moving mean; optionally accumulate trend
__global__ void decomp_kernel(const float* __restrict__ x, float* __restrict__ season,
                              float* __restrict__ trend_acc,
                              const float* __restrict__ w, const float* __restrict__ bb) {
    long i = (long)blockIdx.x * 256 + threadIdx.x;
    if (i >= BLD) return;
    int d = (int)(i % Dd);
    long bl = i / Dd;
    int l = (int)(bl % Ls);
    int b = (int)(bl / Ls);
    const float* base = x + ((long)b * Ls) * Dd + d;
    float xv = x[i];
    float s12 = 0.0f;
    for (int j = l - 5; j <= l + 6; ++j)  { int jj = clampi(j, 0, Ls - 1); s12 += base[(long)jj * Dd]; }
    float s24 = 0.0f;
    for (int j = l - 11; j <= l + 12; ++j){ int jj = clampi(j, 0, Ls - 1); s24 += base[(long)jj * Dd]; }
    float m0 = s12 * (1.0f / 12.0f), m1 = s24 * (1.0f / 24.0f);
    float z0 = xv * w[0] + bb[0], z1 = xv * w[1] + bb[1];
    float zm = fmaxf(z0, z1);
    float e0 = expf(z0 - zm), e1 = expf(z1 - zm);
    float inv = 1.0f / (e0 + e1);
    float mean = (m0 * e0 + m1 * e1) * inv;
    season[i] = xv - mean;
    if (trend_acc) trend_acc[i] += mean;
}

// _my_layernorm part 1: standard LN over D
__global__ __launch_bounds__(256) void ln_kernel(const float* __restrict__ x,
                                                 const float* __restrict__ g,
                                                 const float* __restrict__ bb,
                                                 float* __restrict__ y) {
    int row = blockIdx.x;  // (b,l)
    int t = threadIdx.x;
    const float* xr = x + (long)row * Dd;
    float v0 = xr[t], v1 = xr[t + 256];
    __shared__ float sm[256];
    sm[t] = v0 + v1; __syncthreads();
    for (int s = 128; s > 0; s >>= 1) { if (t < s) sm[t] += sm[t + s]; __syncthreads(); }
    float mu = sm[0] * (1.0f / Dd); __syncthreads();
    float d0 = v0 - mu, d1 = v1 - mu;
    sm[t] = d0 * d0 + d1 * d1; __syncthreads();
    for (int s = 128; s > 0; s >>= 1) { if (t < s) sm[t] += sm[t + s]; __syncthreads(); }
    float rinv = rsqrtf(sm[0] * (1.0f / Dd) + 1e-5f);
    y[(long)row * Dd + t]       = d0 * rinv * g[t] + bb[t];
    y[(long)row * Dd + t + 256] = d1 * rinv * g[t + 256] + bb[t + 256];
}

// _my_layernorm part 2: subtract per-(b,d) mean over L
__global__ void sub_lmean_kernel(const float* __restrict__ in, float* __restrict__ out) {
    int i = blockIdx.x * 256 + threadIdx.x;
    if (i >= Bb * Dd) return;
    int d = i % Dd, b = i / Dd;
    const float* p = in + (long)b * Ls * Dd + d;
    float s = 0.0f;
    for (int l = 0; l < Ls; ++l) s += p[(long)l * Dd];
    s *= (1.0f / Ls);
    float* q = out + (long)b * Ls * Dd + d;
    for (int l = 0; l < Ls; ++l) q[(long)l * Dd] = p[(long)l * Dd] - s;
}

// dec_out = trend_init + circ_conv3(trend_sum, Wt) + seasonal
__global__ void final_kernel(const float* __restrict__ trend, const float* __restrict__ Wt,
                             const float* __restrict__ xmean, const float* __restrict__ seasonal,
                             float* __restrict__ out) {
    int i = blockIdx.x * 256 + threadIdx.x;
    if (i >= Bb * Ls * COUT) return;
    int c = i % COUT, l = (i / COUT) % Ls, b = i / (COUT * Ls);
    const float* Wc = Wt + (long)c * Dd * 3;
    float acc = 0.0f;
    for (int j = 0; j < 3; ++j) {
        int lm = l + j - 1;
        if (lm < 0) lm += Ls; else if (lm >= Ls) lm -= Ls;
        const float* tr = trend + ((long)b * Ls + lm) * Dd;
        for (int d = 0; d < Dd; ++d) acc += tr[d] * Wc[d * 3 + j];
    }
    acc += xmean[b * CIN + c] + seasonal[((long)b * Ls + l) * COUT + c];
    out[i] = acc;
}

// ---------------------------------------------------------------------------
// Host helpers
// ---------------------------------------------------------------------------
static inline dim3 g1(long n) { return dim3((unsigned int)((n + 255) / 256)); }

static void run_gemm(hipStream_t s, const float* X, const float* W, const float* bias,
                     const float* resid, float* Y, int M, int N, int K, int act,
                     unsigned short* abf, unsigned short* wbf, bool cvt_act = true) {
    const int Npad = (N + 63) & ~63;
    long nx = (long)M * K, nw = (long)Npad * K;
    if (cvt_act) cvt_bf16_kernel<<<g1(nx), 256, 0, s>>>(X, abf, nx);
    cvt_bf16_pad_kernel<<<g1(nw), 256, 0, s>>>(W, wbf, N, K, Npad);
    const int nsup = Npad >> 6;
    long tiles = (long)(M / 16) * nsup;
    gemm_wmma_kernel<<<dim3((unsigned int)((tiles + 7) / 8)), 256, 0, s>>>(
        abf, wbf, bias, resid, Y, M, N, K, act);
}

extern "C" void kernel_launch(void* const* d_in, const int* in_sizes, int n_in,
                              void* d_out, int out_size, void* d_ws, size_t ws_size,
                              hipStream_t stream) {
    (void)in_sizes; (void)n_in; (void)out_size; (void)ws_size;
    const float* x           = (const float*)d_in[0];
    const float* emb_enc_w   = (const float*)d_in[1];
    const float* emb_dec_w   = (const float*)d_in[2];
    const float* enc_proj_w  = (const float*)d_in[3];
    const float* enc_proj_b  = (const float*)d_in[4];
    const float* enc_sig_w   = (const float*)d_in[5];
    const float* enc_sig_b   = (const float*)d_in[6];
    const float* enc_four_wr = (const float*)d_in[7];
    const float* enc_four_wi = (const float*)d_in[8];
    const float* enc_ffn_w1  = (const float*)d_in[9];
    const float* enc_ffn_w2  = (const float*)d_in[10];
    const float* enc_dcmp_w  = (const float*)d_in[11];
    const float* enc_dcmp_b  = (const float*)d_in[12];
    const float* enc_norm_g  = (const float*)d_in[13];
    const float* enc_norm_b  = (const float*)d_in[14];
    const float* dec_proj_w  = (const float*)d_in[15];
    const float* dec_proj_b  = (const float*)d_in[16];
    const float* dec_four_wr = (const float*)d_in[17];
    const float* dec_four_wi = (const float*)d_in[18];
    const float* crs_proj_w  = (const float*)d_in[19];
    const float* crs_proj_b  = (const float*)d_in[20];
    const float* crs_four_wr = (const float*)d_in[21];
    const float* crs_four_wi = (const float*)d_in[22];
    const float* dec_ffn_w1  = (const float*)d_in[23];
    const float* dec_ffn_w2  = (const float*)d_in[24];
    const float* dec_dcmp_w  = (const float*)d_in[25];
    const float* dec_dcmp_b  = (const float*)d_in[26];
    const float* dec_trend_w = (const float*)d_in[27];
    const float* dec_norm_g  = (const float*)d_in[28];
    const float* dec_norm_b  = (const float*)d_in[29];
    const float* out_proj_w  = (const float*)d_in[30];
    const float* out_proj_b  = (const float*)d_in[31];

    float* out = (float*)d_out;
    float* out_dec    = out;                                    // (B,L,COUT)
    float* out_series = out + (long)Bb * Ls * COUT;             // (2,B,H,L,L)
    float* out_prior  = out_series + 2L * Bb * Hh * Ls * Ls;    // (2,B,H,L,L)
    float* out_sigma  = out_prior  + 2L * Bb * Hh * Ls * Ls;    // (2,B,H,L)

    // ---- workspace carve (bump allocator, 256B aligned) ----
    char* ws = (char*)d_ws; size_t off = 0;
    auto alloc = [&](size_t bytes) -> char* {
        char* p = ws + off; off = (off + bytes + 255) & ~(size_t)255; return p;
    };
    float* enc   = (float*)alloc(BLD * 4);
    float* decb  = (float*)alloc(BLD * 4);
    float* qb    = (float*)alloc(BLD * 4);
    float* kb    = (float*)alloc(BLD * 4);
    float* attn  = (float*)alloc(BLD * 4);
    float* tmp1  = (float*)alloc(BLD * 4);
    float* tmp2  = (float*)alloc(BLD * 4);
    float* trend = (float*)alloc(BLD * 4);
    long FSZ = (long)Bb * Hh * Ee * MODES * 2;
    float* qf    = (float*)alloc(FSZ * 4);
    float* kf    = (float*)alloc(FSZ * 4);
    float* sel   = (float*)alloc(FSZ * 4);
    float* qkb   = (float*)alloc((long)Bb * Hh * MODES * MODES * 2 * 4);
    float2* trig = (float2*)alloc((long)MODES * Ls * sizeof(float2));
    float* xmean = (float*)alloc((long)Bb * CIN * 4);
    unsigned short* abf = (unsigned short*)alloc((long)M_ROWS * Dd * 2);
    unsigned short* kbf = (unsigned short*)alloc((long)M_ROWS * Dd * 2);
    unsigned short* wbf = (unsigned short*)alloc((long)Dd * Dd * 2);

    const long DD = (long)Dd * Dd;
    const long FQ = (long)Bb * Hh * Ee * MODES;
    const long SC_TILES = (long)Bb * Hh * 7 * 2;   // scores wave-tiles

    trig_kernel<<<g1(MODES * Ls), 256, 0, stream>>>(trig);
    colmean_kernel<<<g1(Bb * CIN), 256, 0, stream>>>(x, xmean);

    // ======================= encoder =======================
    embed_kernel<<<g1(BLD), 256, 0, stream>>>(x, emb_enc_w, nullptr, enc);
    for (int layer = 0; layer < 2; ++layer) {
        const float* Wp = enc_proj_w + (long)layer * 4 * DD;
        const float* bp = enc_proj_b + (long)layer * 4 * Dd;
        // Q and K projections share one bf16 conversion of enc
        run_gemm(stream, enc, Wp + 0 * DD, bp + 0 * Dd, nullptr, qb, M_ROWS, Dd, Dd, 0, abf, wbf, true);
        run_gemm(stream, enc, Wp + 1 * DD, bp + 1 * Dd, nullptr, kb, M_ROWS, Dd, Dd, 0, abf, wbf, false);
        sigma_kernel<<<g1(Bb * Hh * Ls), 256, 0, stream>>>(
            enc, enc_sig_w + (long)layer * Hh * Dd, enc_sig_b + layer * Hh,
            out_sigma + (long)layer * Bb * Hh * Ls);
        prior_kernel<<<g1((long)Bb * Hh * Ls * Ls), 256, 0, stream>>>(
            out_sigma + (long)layer * Bb * Hh * Ls,
            out_prior + (long)layer * Bb * Hh * Ls * Ls);
        // attention series via batched WMMA scores + in-place softmax
        {
            float* Sl = out_series + (long)layer * Bb * Hh * Ls * Ls;
            cvt_bf16_kernel<<<g1(BLD), 256, 0, stream>>>(qb, abf, BLD);
            cvt_bf16_kernel<<<g1(BLD), 256, 0, stream>>>(kb, kbf, BLD);
            scores_wmma_kernel<<<dim3((unsigned int)((SC_TILES + 7) / 8)), 256, 0, stream>>>(
                abf, kbf, Sl);
            softmax_inplace_kernel<<<dim3(Bb * Hh * Ls), 128, 0, stream>>>(Sl);
        }
        // FourierBlock on q
        dft_fwd_kernel<<<g1(FQ), 256, 0, stream>>>(qb, trig, qf);
        mode_gemm_kernel<<<g1(FQ), 256, 0, stream>>>(qf, enc_four_wr, enc_four_wi, sel);
        idft_kernel<<<g1(BLD), 256, 0, stream>>>(sel, trig, attn, 1.0f / (float)Ls);
        // attn out proj + residual(enc)
        run_gemm(stream, attn, Wp + 3 * DD, bp + 3 * Dd, enc, tmp1, M_ROWS, Dd, Dd, 0, abf, wbf);
        decomp_kernel<<<g1(BLD), 256, 0, stream>>>(tmp1, tmp2, nullptr,
            enc_dcmp_w + (layer * 2 + 0) * 2, enc_dcmp_b + (layer * 2 + 0) * 2);
        // FFN (GELU fused in epilogue of first GEMM)
        run_gemm(stream, tmp2, enc_ffn_w1 + (long)layer * DFF * Dd, nullptr, nullptr,
                 qb, M_ROWS, DFF, Dd, 1, abf, wbf);
        run_gemm(stream, qb, enc_ffn_w2 + (long)layer * Dd * DFF, nullptr, tmp2,
                 tmp1, M_ROWS, Dd, DFF, 0, abf, wbf);
        decomp_kernel<<<g1(BLD), 256, 0, stream>>>(tmp1, enc, nullptr,
            enc_dcmp_w + (layer * 2 + 1) * 2, enc_dcmp_b + (layer * 2 + 1) * 2);
    }
    ln_kernel<<<dim3((unsigned int)M_ROWS), 256, 0, stream>>>(enc, enc_norm_g, enc_norm_b, tmp1);
    sub_lmean_kernel<<<g1(Bb * Dd), 256, 0, stream>>>(tmp1, enc);   // final encoder state

    // ======================= decoder =======================
    embed_kernel<<<g1(BLD), 256, 0, stream>>>(x, emb_dec_w, xmean, decb);  // seasonal_init embed
    run_gemm(stream, decb, dec_proj_w + 0 * DD, dec_proj_b + 0 * Dd, nullptr,
             qb, M_ROWS, Dd, Dd, 0, abf, wbf);
    dft_fwd_kernel<<<g1(FQ), 256, 0, stream>>>(qb, trig, qf);
    mode_gemm_kernel<<<g1(FQ), 256, 0, stream>>>(qf, dec_four_wr, dec_four_wi, sel);
    idft_kernel<<<g1(BLD), 256, 0, stream>>>(sel, trig, attn, 1.0f / (float)Ls);
    run_gemm(stream, attn, dec_proj_w + 3 * DD, dec_proj_b + 3 * Dd, decb,
             tmp1, M_ROWS, Dd, Dd, 0, abf, wbf);
    zero_kernel<<<g1(BLD), 256, 0, stream>>>(trend, BLD);
    decomp_kernel<<<g1(BLD), 256, 0, stream>>>(tmp1, tmp2, trend,
        dec_dcmp_w + 0 * 2, dec_dcmp_b + 0 * 2);                       // h, t1
    // cross attention (Fourier)
    run_gemm(stream, tmp2, crs_proj_w + 0 * DD, crs_proj_b + 0 * Dd, nullptr,
             qb, M_ROWS, Dd, Dd, 0, abf, wbf);                         // qc
    run_gemm(stream, enc, crs_proj_w + 1 * DD, crs_proj_b + 1 * Dd, nullptr,
             kb, M_ROWS, Dd, Dd, 0, abf, wbf);                         // kc
    dft_fwd_kernel<<<g1(FQ), 256, 0, stream>>>(qb, trig, qf);
    dft_fwd_kernel<<<g1(FQ), 256, 0, stream>>>(kb, trig, kf);
    cross_qk_kernel<<<g1((long)Bb * Hh * MODES * MODES), 256, 0, stream>>>(qf, kf, qkb);
    cross_qkv_kernel<<<g1(FQ), 256, 0, stream>>>(qkb, kf, sel);
    mode_gemm_kernel<<<g1(FQ), 256, 0, stream>>>(sel, crs_four_wr, crs_four_wi, qf);
    idft_kernel<<<g1(BLD), 256, 0, stream>>>(qf, trig, attn,
        1.0f / ((float)Ls * (float)Dd * (float)Dd));
    run_gemm(stream, attn, crs_proj_w + 3 * DD, crs_proj_b + 3 * Dd, tmp2,
             tmp1, M_ROWS, Dd, Dd, 0, abf, wbf);
    decomp_kernel<<<g1(BLD), 256, 0, stream>>>(tmp1, qb, trend,
        dec_dcmp_w + 1 * 2, dec_dcmp_b + 1 * 2);                       // h, t2
    // decoder FFN
    run_gemm(stream, qb, dec_ffn_w1, nullptr, nullptr, kb, M_ROWS, DFF, Dd, 1, abf, wbf);
    run_gemm(stream, kb, dec_ffn_w2, nullptr, qb, tmp1, M_ROWS, Dd, DFF, 0, abf, wbf);
    decomp_kernel<<<g1(BLD), 256, 0, stream>>>(tmp1, tmp2, trend,
        dec_dcmp_w + 2 * 2, dec_dcmp_b + 2 * 2);                       // h, t3
    // final norm + seasonal projection + trend conv
    ln_kernel<<<dim3((unsigned int)M_ROWS), 256, 0, stream>>>(tmp2, dec_norm_g, dec_norm_b, tmp1);
    sub_lmean_kernel<<<g1(Bb * Dd), 256, 0, stream>>>(tmp1, tmp2);
    run_gemm(stream, tmp2, out_proj_w, out_proj_b, nullptr, kb, M_ROWS, COUT, Dd, 0, abf, wbf);
    final_kernel<<<g1((long)Bb * Ls * COUT), 256, 0, stream>>>(trend, dec_trend_w, xmean,
                                                               kb, out_dec);
}
